// ELGCA_EncoderBlock_64063732187360
// MI455X (gfx1250) — matrix operand order repaired and checked by reading, not compiled, not run on hardware
//
#include <hip/hip_runtime.h>
#include <hip/hip_bf16.h>
#include <math.h>

// ---------------------------------------------------------------------------
// ELGCA encoder block for MI455X (gfx1250, wave32, WMMA).
//
// The three 1x1 convs (qkvl 64->128, fc1 128->512, fc2 512->128) carry ~96%
// of the ~38 GFLOP and run on v_wmma_f32_16x16x32_bf16 (f32 accumulate).
//
// Data movement strategy:
//  * All intermediate activations are NHWC (channels innermost), so a WMMA
//    B fragment (n = lane&15, k = e + 16*(lane>>4)) is 16 consecutive bf16.
//  * Each block stages its 64-pixel x KCH activation panel into LDS ONCE
//    (coalesced uint4 loads, every element fetched once per block); all 8
//    waves then feed B fragments from LDS via conflict-free ds_load_b128
//    (row pitch KCH+8 => half-wave lanes hit banks 4n mod 64, all distinct).
//  * Weights are pre-packed to bf16 in exact A-fragment order -> one
//    contiguous v16bf load per fragment, zero cvt in the hot loop.
//  * C/D layout (m = j + 8*(lane>>4)) gives 8 consecutive output channels
//    -> single 16B v8bf store per tile.
//
// Workspace requirement: ~412 MB.
// ---------------------------------------------------------------------------

typedef __attribute__((ext_vector_type(16))) __bf16 v16bf;
typedef __attribute__((ext_vector_type(8)))  __bf16 v8bf;
typedef __attribute__((ext_vector_type(8)))  float  v8f;

#define BATCH 32
#define CDIM  128
#define HIMG  64
#define WIMG  64
#define HW    4096   // 64*64
#define HWQ   1024   // 32*32 (pooled)

__device__ __forceinline__ float gelu_exact(float v) {
    return 0.5f * v * (1.0f + erff(v * 0.7071067811865476f));
}

// ---------------------------------------------------------------------------
// Pack f32 weights (M, KCH) into bf16 A-fragments.
// ap[((mtile*(KCH/32) + ktile)*32 + lane)*16 + e], contiguous-k A mapping (used here for A and B):
// m = mtile*16 + (lane&15), k = ktile*32 + 16*(lane>>4) + e.
// ---------------------------------------------------------------------------
__global__ void pack_w_kernel(const float* __restrict__ w,
                              __bf16* __restrict__ ap, int M, int KCH) {
    long idx = (long)blockIdx.x * blockDim.x + threadIdx.x;
    long total = (long)(M / 16) * (KCH / 32) * 32 * 16;
    if (idx >= total) return;
    int e    = (int)(idx & 15);
    int lane = (int)((idx >> 4) & 31);
    long tile = idx >> 9;
    int kt = (int)(tile % (KCH / 32));
    int mt = (int)(tile / (KCH / 32));
    int m = mt * 16 + (lane & 15);
    int k = kt * 32 + (lane >> 4) * 16 + e;
    ap[idx] = (__bf16)w[(long)m * KCH + k];
}

// ---------------------------------------------------------------------------
// LN1: reads x in NCHW f32 (coalesced across pixel-threads), writes NHWC bf16.
// Reference LN does NOT subtract the mean from x (mean only feeds variance).
// ---------------------------------------------------------------------------
__global__ void ln1_bf16_kernel(const float* __restrict__ x,
                                const float* __restrict__ w,
                                __bf16* __restrict__ out) {
    int p = blockIdx.x * blockDim.x + threadIdx.x;
    int b = blockIdx.y;
    if (p >= HW) return;
    const float* xb = x + (long)b * CDIM * HW + p;
    float s = 0.f, ss = 0.f;
    for (int c = 0; c < CDIM; ++c) {
        float v = xb[(long)c * HW];
        s += v; ss += v * v;
    }
    float mu  = s * (1.0f / CDIM);
    float var = ss * (1.0f / CDIM) - mu * mu;
    float inv = rsqrtf(var + 1e-5f);
    __bf16* ob = out + ((long)b * HW + p) * CDIM;
    for (int c0 = 0; c0 < CDIM; c0 += 8) {
        v8bf pk;
        #pragma unroll
        for (int j = 0; j < 8; ++j)
            pk[j] = (__bf16)(xb[(long)(c0 + j) * HW] * inv * w[c0 + j]);
        *(v8bf*)(ob + c0) = pk;
    }
}

// LN2: reads NHWC f32 residual buffer (contiguous per thread), writes NHWC bf16.
__global__ void ln2_bf16_kernel(const float* __restrict__ xn,   // (B,HW,128)
                                const float* __restrict__ w,
                                __bf16* __restrict__ out) {
    int p = blockIdx.x * blockDim.x + threadIdx.x;
    int b = blockIdx.y;
    if (p >= HW) return;
    const float* xb = xn + ((long)b * HW + p) * CDIM;
    float s = 0.f, ss = 0.f;
    for (int c = 0; c < CDIM; ++c) {
        float v = xb[c];
        s += v; ss += v * v;
    }
    float mu  = s * (1.0f / CDIM);
    float var = ss * (1.0f / CDIM) - mu * mu;
    float inv = rsqrtf(var + 1e-5f);
    __bf16* ob = out + ((long)b * HW + p) * CDIM;
    for (int c0 = 0; c0 < CDIM; c0 += 8) {
        v8bf pk;
        #pragma unroll
        for (int j = 0; j < 8; ++j)
            pk[j] = (__bf16)(xb[c0 + j] * inv * w[c0 + j]);
        *(v8bf*)(ob + c0) = pk;
    }
}

// ---------------------------------------------------------------------------
// Depthwise 3x3 (pad 1) + GELU on NHWC bf16, channel-fastest threads.
// SELF=false: out_f32(NHWC,pitch 128) = gelu(conv(in)) + res_x(NCHW)   (x1)
// SELF=true : out_bf (NHWC,pitch CH)  = in + gelu(conv(in))            (pos)
// ---------------------------------------------------------------------------
template<bool SELF>
__global__ void dwconv3_kernel(const __bf16* __restrict__ in,  // NHWC, pitch inP
                               const float* __restrict__ w,    // (CH,1,3,3)
                               const float* __restrict__ bias, // (CH)
                               const float* __restrict__ resx, // x NCHW (!SELF)
                               float* __restrict__ out_f32,    // NHWC p128 (!SELF)
                               __bf16* __restrict__ out_bf,    // NHWC pCH (SELF)
                               int CH, int inP, long total) {
    long idx = (long)blockIdx.x * blockDim.x + threadIdx.x;
    if (idx >= total) return;
    int c  = (int)(idx % CH);
    long pr = idx / CH;
    int xq = (int)(pr & 63);
    int yq = (int)((pr >> 6) & 63);
    int b  = (int)(pr >> 12);

    const __bf16* ib = in + (long)b * HW * inP + c;
    const float*  wc = w + c * 9;
    float acc = bias[c];
    #pragma unroll
    for (int ky = 0; ky < 3; ++ky) {
        int yy = yq + ky - 1;
        if (yy < 0 || yy >= HIMG) continue;
        #pragma unroll
        for (int kx = 0; kx < 3; ++kx) {
            int xx = xq + kx - 1;
            if (xx < 0 || xx >= WIMG) continue;
            acc += (float)ib[(long)(yy * WIMG + xx) * inP] * wc[ky * 3 + kx];
        }
    }
    float g = gelu_exact(acc);
    long p = (long)yq * WIMG + xq;
    if (SELF) {
        out_bf[((long)b * HW + p) * CH + c] =
            (__bf16)((float)ib[p * inP] + g);
    } else {
        out_f32[((long)b * HW + p) * CDIM + c] =
            g + resx[((long)b * CDIM + c) * HW + p];
    }
}

// ---------------------------------------------------------------------------
// 1x1 conv WMMA GEMM, LDS-staged.
// Block = 256 threads = 8 waves; block tile = 128 M rows x 64 pixels.
// Grid: (HW/64, M/128, B). The 64-pixel x KCH activation panel is loaded to
// LDS once per block; each wave feeds 4 accumulators (16Mx64N) from it.
// ---------------------------------------------------------------------------
template<int KCH, bool GELU_ACT, bool FC2>
__global__ __launch_bounds__(256)
void gemm1x1_kernel(const __bf16* __restrict__ act,  // (B,HW,kstride), +chan offset
                    const __bf16* __restrict__ ap,   // packed bf16 weights
                    const float* __restrict__ bias,  // (M)
                    const float* __restrict__ res,   // (B,HW,128) f32, FC2 only
                    __bf16* __restrict__ outb,       // (B,HW,M) bf16, !FC2
                    float* __restrict__ outf,        // (B,128,HW) NCHW f32, FC2
                    int M, int kstride) {
    constexpr int LP = KCH + 8;          // LDS row pitch (bf16): conflict-free
    extern __shared__ __bf16 Bs[];       // [64][LP]

    const int lane  = threadIdx.x & 31;
    const int wave  = threadIdx.x >> 5;
    const int b     = blockIdx.z;
    const int nbase = blockIdx.x * 64;
    const int mtile = blockIdx.y * 8 + wave;
    const int mrow  = mtile * 16;

    const int bn  = lane & 15;           // B/C column
    const int bko = (lane >> 4) * 16;    // B fragment k offset

    const __bf16* abase = ap + ((long)mtile * (KCH / 32) * 32 + lane) * 16;
    const __bf16* actb  = act + (long)b * HW * kstride;

    // Cooperative fill: each global element fetched exactly once per block.
    constexpr int CHUNKS = 64 * KCH / 8;          // 16B chunks
    #pragma unroll
    for (int i = threadIdx.x; i < CHUNKS; i += 256) {
        int n  = i / (KCH / 8);
        int kc = i - n * (KCH / 8);
        uint4 v = *(const uint4*)(actb + (long)(nbase + n) * kstride + kc * 8);
        *(uint4*)(&Bs[n * LP + kc * 8]) = v;
    }
    __syncthreads();

    v8f acc[4];
    #pragma unroll
    for (int t = 0; t < 4; ++t)
        acc[t] = (v8f){0.f, 0.f, 0.f, 0.f, 0.f, 0.f, 0.f, 0.f};

    #pragma unroll
    for (int kt = 0; kt < KCH / 32; ++kt) {
        v16bf A = *(const v16bf*)(abase + (long)kt * 512);
        #pragma unroll
        for (int t = 0; t < 4; ++t) {
            const __bf16* bp = &Bs[(t * 16 + bn) * LP + kt * 32 + bko];
            v8bf lo = *(const v8bf*)(bp);
            v8bf hi = *(const v8bf*)(bp + 8);
            v16bf Bm = __builtin_shufflevector(lo, hi,
                0, 1, 2, 3, 4, 5, 6, 7, 8, 9, 10, 11, 12, 13, 14, 15);
            acc[t] = __builtin_amdgcn_wmma_f32_16x16x32_bf16(
                false, A, false, Bm, (short)0, acc[t], false, false);
        }
    }

    // Epilogue: m = mrow + 8*(lane>>4) + j  -> 8 consecutive channels.
    const int m0 = mrow + ((lane >> 4) << 3);
    #pragma unroll
    for (int t = 0; t < 4; ++t) {
        int n = nbase + t * 16 + bn;
        if (!FC2) {
            v8bf pk;
            #pragma unroll
            for (int j = 0; j < 8; ++j) {
                float v = acc[t][j] + bias[m0 + j];
                if (GELU_ACT) v = gelu_exact(v);
                pk[j] = (__bf16)v;
            }
            *(v8bf*)(outb + ((long)b * HW + n) * M + m0) = pk;
        } else {
            v8f r = *(const v8f*)(res + ((long)b * HW + n) * CDIM + m0);
            #pragma unroll
            for (int j = 0; j < 8; ++j) {
                float v = acc[t][j] + bias[m0 + j] + r[j];
                outf[((long)b * CDIM + m0 + j) * HW + n] = v;   // NCHW output
            }
        }
    }
}

// ---------------------------------------------------------------------------
// Pooling on NHWC x2 (B,HW,128); head h = channels [32h, 32h+32).
// Channel-fastest threads; outputs NHWC-pooled (B,HWQ,32).
// ---------------------------------------------------------------------------
__global__ void avgpool_kernel(const __bf16* __restrict__ x2,
                               float* __restrict__ qf) {
    long idx = (long)blockIdx.x * blockDim.x + threadIdx.x;
    if (idx >= (long)BATCH * HWQ * 32) return;
    int c  = (int)(idx & 31);
    int ox = (int)((idx >> 5) & 31);
    int oy = (int)((idx >> 10) & 31);
    int b  = (int)(idx >> 15);
    const __bf16* ib = x2 + (long)b * HW * CDIM + c;     // head 0 (q)
    float s = 0.f;
    #pragma unroll
    for (int ky = 0; ky < 3; ++ky) {
        int iy = 2 * oy - 1 + ky;
        if (iy < 0 || iy >= HIMG) continue;
        #pragma unroll
        for (int kx = 0; kx < 3; ++kx) {
            int ix = 2 * ox - 1 + kx;
            if (ix < 0 || ix >= WIMG) continue;
            s += (float)ib[(long)(iy * WIMG + ix) * CDIM];
        }
    }
    qf[((long)b * HWQ + oy * 32 + ox) * 32 + c] = s * (1.0f / 9.0f);
}

__global__ void maxpool_kernel(const __bf16* __restrict__ x2,
                               float* __restrict__ kf) {
    long idx = (long)blockIdx.x * blockDim.x + threadIdx.x;
    if (idx >= (long)BATCH * HWQ * 32) return;
    int c  = (int)(idx & 31);
    int ox = (int)((idx >> 5) & 31);
    int oy = (int)((idx >> 10) & 31);
    int b  = (int)(idx >> 15);
    const __bf16* ib = x2 + (long)b * HW * CDIM + 32 + c;  // head 1 (k)
    int iy = 2 * oy, ix = 2 * ox;
    float m =        (float)ib[(long)(iy * WIMG + ix) * CDIM];
    m = fmaxf(m,     (float)ib[(long)(iy * WIMG + ix + 1) * CDIM]);
    m = fmaxf(m,     (float)ib[(long)((iy + 1) * WIMG + ix) * CDIM]);
    m = fmaxf(m,     (float)ib[(long)((iy + 1) * WIMG + ix + 1) * CDIM]);
    kf[((long)b * HWQ + oy * 32 + ox) * 32 + c] = m;
}

// qk[b,i,j] = sum_n qf[b,n,i] * kf[b,n,j]   (n = 1024; tiny op)
__global__ void qk_kernel(const float* __restrict__ qf,
                          const float* __restrict__ kf,
                          float* __restrict__ qk) {
    int j = threadIdx.x, i = threadIdx.y, b = blockIdx.x;
    const float* q = qf + (long)b * HWQ * 32 + i;
    const float* k = kf + (long)b * HWQ * 32 + j;
    float s = 0.f;
    for (int n = 0; n < HWQ; ++n) s += q[n * 32] * k[n * 32];
    qk[(long)b * 1024 + i * 32 + j] = s;
}

// sm[b,i,j] = softmax over i of qk[b,i,j]
__global__ void softmax_axis1_kernel(const float* __restrict__ qk,
                                     float* __restrict__ sm) {
    int j = threadIdx.x, b = blockIdx.x;
    const float* col = qk + (long)b * 1024 + j;
    float mx = -3.4e38f;
    for (int i = 0; i < 32; ++i) mx = fmaxf(mx, col[i * 32]);
    float s = 0.f;
    for (int i = 0; i < 32; ++i) s += __expf(col[i * 32] - mx);
    float inv = 1.0f / s;
    for (int i = 0; i < 32; ++i)
        sm[(long)b * 1024 + i * 32 + j] = __expf(col[i * 32] - mx) * inv;
}

// outbuf[b,n,96+c] = x[b,96+c,n] + sum_d sm[b,d,c] * v[b,n,64+d]
__global__ void attnout_kernel(const float* __restrict__ sm,
                               const __bf16* __restrict__ x2,
                               const float* __restrict__ x,
                               float* __restrict__ outbuf) {
    long idx = (long)blockIdx.x * blockDim.x + threadIdx.x;
    if (idx >= (long)BATCH * HW * 32) return;
    int c = (int)(idx & 31);
    int n = (int)((idx >> 5) & (HW - 1));
    int b = (int)(idx >> 17);
    const __bf16* vb = x2 + ((long)b * HW + n) * CDIM + 64;
    const float*  sb = sm + (long)b * 1024 + c;
    float s = 0.f;
    #pragma unroll 4
    for (int d = 0; d < 32; ++d) s += sb[d * 32] * (float)vb[d];
    outbuf[((long)b * HW + n) * CDIM + 96 + c] =
        s + x[((long)b * CDIM + 96 + c) * HW + n];
}

// outbuf[b,n,64+c] = lfeat (head 3) + x residual
__global__ void lfeat_kernel(const __bf16* __restrict__ x2,
                             const float* __restrict__ x,
                             float* __restrict__ outbuf) {
    long idx = (long)blockIdx.x * blockDim.x + threadIdx.x;
    if (idx >= (long)BATCH * HW * 32) return;
    int c = (int)(idx & 31);
    int n = (int)((idx >> 5) & (HW - 1));
    int b = (int)(idx >> 17);
    outbuf[((long)b * HW + n) * CDIM + 64 + c] =
        (float)x2[((long)b * HW + n) * CDIM + 96 + c] +
        x[((long)b * CDIM + 64 + c) * HW + n];
}

// ---------------------------------------------------------------------------
extern "C" void kernel_launch(void* const* d_in, const int* in_sizes, int n_in,
                              void* d_out, int out_size, void* d_ws, size_t ws_size,
                              hipStream_t stream) {
    const float* x      = (const float*)d_in[0];
    const float* ln1_w  = (const float*)d_in[1];
    const float* ln2_w  = (const float*)d_in[2];
    const float* dw_w   = (const float*)d_in[3];
    const float* dw_b   = (const float*)d_in[4];
    const float* qkvl_w = (const float*)d_in[5];
    const float* qkvl_b = (const float*)d_in[6];
    const float* fc1_w  = (const float*)d_in[7];
    const float* fc1_b  = (const float*)d_in[8];
    const float* pos_w  = (const float*)d_in[9];
    const float* pos_b  = (const float*)d_in[10];
    const float* fc2_w  = (const float*)d_in[11];
    const float* fc2_b  = (const float*)d_in[12];
    float* out = (float*)d_out;

    // Workspace layout (~412 MB); all sizes are 256B multiples.
    char* ws = (char*)d_ws;
    size_t off = 0;
    auto alloc = [&](size_t bytes) { char* p = ws + off; off += bytes; return p; };
    __bf16* hbf    = (__bf16*)alloc((size_t)BATCH * HW * CDIM * 2);   // NHWC bf16
    float*  outbuf = (float*) alloc((size_t)BATCH * HW * CDIM * 4);   // NHWC f32
    __bf16* x2bf   = (__bf16*)alloc((size_t)BATCH * HW * CDIM * 2);   // NHWC bf16
    float*  qf     = (float*) alloc((size_t)BATCH * HWQ * 32 * 4);
    float*  kf     = (float*) alloc((size_t)BATCH * HWQ * 32 * 4);
    float*  qkm    = (float*) alloc((size_t)BATCH * 32 * 32 * 4);
    float*  smm    = (float*) alloc((size_t)BATCH * 32 * 32 * 4);
    __bf16* h1bf   = (__bf16*)alloc((size_t)BATCH * HW * 512 * 2);    // NHWC bf16
    __bf16* h2bf   = (__bf16*)alloc((size_t)BATCH * HW * 512 * 2);    // NHWC bf16
    __bf16* apQ    = (__bf16*)alloc((size_t)128 * 64 * 2);            // packed qkvl_w
    __bf16* apF1   = (__bf16*)alloc((size_t)512 * 128 * 2);           // packed fc1_w
    __bf16* apF2   = (__bf16*)alloc((size_t)128 * 512 * 2);           // packed fc2_w
    (void)ws_size; (void)in_sizes; (void)n_in; (void)out_size;

    // 0) Pack weights into WMMA A-fragment order (bf16).
    pack_w_kernel<<<(128 * 64 + 255) / 256, 256, 0, stream>>>(qkvl_w, apQ, 128, 64);
    pack_w_kernel<<<(512 * 128 + 255) / 256, 256, 0, stream>>>(fc1_w, apF1, 512, 128);
    pack_w_kernel<<<(128 * 512 + 255) / 256, 256, 0, stream>>>(fc2_w, apF2, 128, 512);

    // 1) LN1(x) -> NHWC bf16
    ln1_bf16_kernel<<<dim3(HW / 256, BATCH), 256, 0, stream>>>(x, ln1_w, hbf);

    // 2) x1 branch: depthwise 3x3 + gelu + residual(x) -> outbuf ch[0..64)
    {
        long total = (long)BATCH * HW * 64;
        dwconv3_kernel<false><<<(total + 255) / 256, 256, 0, stream>>>(
            hbf, dw_w, dw_b, x, outbuf, nullptr, 64, CDIM, total);
    }

    // 3) qkvl 1x1 (K=64 -> M=128) + gelu, WMMA -> x2bf (NHWC)
    gemm1x1_kernel<64, true, false>
        <<<dim3(HW / 64, 1, BATCH), 256, 64 * (64 + 8) * 2, stream>>>(
            hbf + 64, apQ, qkvl_b, nullptr, x2bf, nullptr, 128, CDIM);

    // 4/5) pooling on q (head 0) and k (head 1)
    {
        long total = (long)BATCH * HWQ * 32;
        avgpool_kernel<<<(total + 255) / 256, 256, 0, stream>>>(x2bf, qf);
        maxpool_kernel<<<(total + 255) / 256, 256, 0, stream>>>(x2bf, kf);
    }

    // 6/7) 32x32 logits + softmax over axis 1
    qk_kernel<<<BATCH, dim3(32, 32), 0, stream>>>(qf, kf, qkm);
    softmax_axis1_kernel<<<BATCH, 32, 0, stream>>>(qkm, smm);

    // 8/9) attention out + residual -> outbuf ch[96..128); lfeat -> ch[64..96)
    {
        long total = (long)BATCH * HW * 32;
        attnout_kernel<<<(total + 255) / 256, 256, 0, stream>>>(smm, x2bf, x, outbuf);
        lfeat_kernel<<<(total + 255) / 256, 256, 0, stream>>>(x2bf, x, outbuf);
    }

    // 10) LN2(outbuf) -> NHWC bf16 (reuse hbf)
    ln2_bf16_kernel<<<dim3(HW / 256, BATCH), 256, 0, stream>>>(outbuf, ln2_w, hbf);

    // 11) fc1 1x1 (K=128 -> M=512) + gelu, WMMA -> h1bf (NHWC)
    gemm1x1_kernel<128, true, false>
        <<<dim3(HW / 64, 512 / 128, BATCH), 256, 64 * (128 + 8) * 2, stream>>>(
            hbf, apF1, fc1_b, nullptr, h1bf, nullptr, 512, CDIM);

    // 12) pos depthwise 3x3: h2 = h1 + gelu(conv(h1)) -> NHWC bf16
    {
        long total = (long)BATCH * HW * 512;
        dwconv3_kernel<true><<<(total + 255) / 256, 256, 0, stream>>>(
            h1bf, pos_w, pos_b, nullptr, nullptr, h2bf, 512, 512, total);
    }

    // 13) fc2 1x1 (K=512 -> M=128), WMMA, fused bias + residual(outbuf) -> NCHW d_out
    gemm1x1_kernel<512, false, true>
        <<<dim3(HW / 64, 1, BATCH), 256, 64 * (512 + 8) * 2, stream>>>(
            h2bf, apF2, fc2_b, outbuf, nullptr, out, 128, 512);
}
